// Model_61598420959695
// MI455X (gfx1250) — matrix operand order repaired and checked by reading, not compiled, not run on hardware
//
#include <hip/hip_runtime.h>
#include <hip/hip_bf16.h>

typedef __attribute__((ext_vector_type(16))) _Float16 v16h;
typedef __attribute__((ext_vector_type(8)))  _Float16 v8h;   // 16B -> b128
typedef __attribute__((ext_vector_type(4)))  _Float16 v4h;   // 8B
typedef __attribute__((ext_vector_type(8)))  float    v8f;
typedef unsigned int u32x4 __attribute__((ext_vector_type(4)));
typedef int          i32x8 __attribute__((ext_vector_type(8)));
typedef int          i32x4 __attribute__((ext_vector_type(4)));

#define B_SZ 256
#define T_SZ 512
#define D_SZ 512
#define A_STRIDE 520   // halves: 512 + 4-DWORD TDM pad -> conflict-free ds_load_b128
#define S_STRIDE 516   // floats: 512 + pad

union Frag { v16h v; v8h h[2]; };

// ---------------------------------------------------------------------------
// Row L2-normalize f32[rows][512] -> f16[rows][512].  One wave per row.
// ---------------------------------------------------------------------------
__global__ void norm_rows_f16(const float* __restrict__ src,
                              _Float16* __restrict__ dst,
                              int nrows) {
  const int lane = threadIdx.x & 31;
  const int wave = threadIdx.x >> 5;
  const int row  = blockIdx.x * 8 + wave;
  if (row >= nrows) return;

  const float4* s = (const float4*)(src + (size_t)row * D_SZ);
  float4 x[4];
  float ss = 0.f;
#pragma unroll
  for (int i = 0; i < 4; ++i) {
    x[i] = s[lane + i * 32];
    ss += x[i].x * x[i].x + x[i].y * x[i].y + x[i].z * x[i].z + x[i].w * x[i].w;
  }
#pragma unroll
  for (int off = 16; off >= 1; off >>= 1)
    ss += __shfl_xor(ss, off, 32);

  const float scale = 1.0f / fmaxf(sqrtf(ss), 1e-8f);

  _Float16* d = dst + (size_t)row * D_SZ;
#pragma unroll
  for (int i = 0; i < 4; ++i) {
    v4h h;
    h.x = (_Float16)(x[i].x * scale);
    h.y = (_Float16)(x[i].y * scale);
    h.z = (_Float16)(x[i].z * scale);
    h.w = (_Float16)(x[i].w * scale);
    *(v4h*)(d + lane * 4 + i * 128) = h;
  }
}

// ---------------------------------------------------------------------------
// Fused per-b GEMM + masked softmax + diagonal argmax/gather.
// Block = 256 thr (8 waves): b = blk>>4, jbase = (blk&15)*16; wave w owns
// t in [w*64, w*64+64).  A tile (16 x 512 f16, contiguous 16KB of sn) is
// TDM-DMA'd into LDS with per-row pad; C tile (16 x 512 f32) staged in LDS
// for the block-wide softmax over t.
// ---------------------------------------------------------------------------
__global__ void __launch_bounds__(256)
sim_fused_wmma(const _Float16* __restrict__ snh,   // [256][512] f16 normalized
               const _Float16* __restrict__ vnh,   // [256*512][512] f16 normalized
               const float* __restrict__ video,    // original f32 video
               const int* __restrict__ lengths,
               float* __restrict__ out)            // sim + video_max tail
{
  __shared__ struct {
    _Float16 A[16 * A_STRIDE];   // MUST be first member: TDM lds_addr = 0
    float    S[16 * S_STRIDE];
  } sm;

  const int lane  = threadIdx.x & 31;
  const int wave  = threadIdx.x >> 5;
  const int b     = blockIdx.x >> 4;
  const int jt    = blockIdx.x & 15;
  const int jbase = jt * 16;
  const int tbase = wave * 64;
  const int lm    = lane & 15;
  const int hi    = lane >> 4;

  // ---- Stage A tile into LDS via Tensor Data Mover --------------------------
  const _Float16* gA = snh + (size_t)jbase * D_SZ;   // contiguous 16 KB
#if __has_builtin(__builtin_amdgcn_tensor_load_to_lds)
  if (wave == 0) {
    const unsigned long long ga = (unsigned long long)(const void*)gA;
    u32x4 g0 = {0u, 0u, 0u, 0u};
    i32x8 g1 = {0, 0, 0, 0, 0, 0, 0, 0};
    i32x4 g2 = {0, 0, 0, 0}, g3 = {0, 0, 0, 0};
    i32x8 g4 = {0, 0, 0, 0, 0, 0, 0, 0};
    g0[0] = 1u;                                   // count=1 (valid), user mode
    g0[1] = 0u;                                   // lds_addr = 0 (sm.A)
    g0[2] = (unsigned)(ga & 0xffffffffull);       // global_addr lo
    g0[3] = (unsigned)((ga >> 32) & 0x01ffffffull) | (2u << 30);  // hi | type=2
    // data_size=2B, pad_enable, pad_interval=256 DW (code 7), pad_amount=4 DW
    g1[0] = (1 << 16) | (1 << 20) | (7 << 22) | (3 << 25);
    g1[1] = (int)((512u & 0xffffu) << 16);        // tensor_dim0 = 512 (lo16)
    g1[2] = (int)((16u & 0xffffu) << 16);         // dim0 hi=0 | tensor_dim1 = 16
    g1[3] = (int)(512u << 16);                    // dim1 hi=0 | tile_dim0 = 512
    g1[4] = 16;                                   // tile_dim1 = 16, tile_dim2 = 0
    g1[5] = 512;                                  // tensor_dim0_stride = 512
    __builtin_amdgcn_tensor_load_to_lds(g0, g1, g2, g3, g4, 0);
    __builtin_amdgcn_s_wait_tensorcnt(0);
  }
#else
  // Fallback: cooperative copy with the same padded layout.
  for (int idx = threadIdx.x; idx < 16 * 64; idx += 256) {   // 8 halves per item
    const int r = idx >> 6, c = (idx & 63) * 8;
    *(v8h*)&sm.A[r * A_STRIDE + c] = *(const v8h*)(gA + r * D_SZ + c);
  }
#endif
  __syncthreads();

  // ---- WMMA K-loop ----------------------------------------------------------
  // A fragment: lanes 0-15 hold K k..k+15, lanes 16-31 hold
  // k+16..k+31 of row m=lm  -> two b128 ds loads.
  const _Float16* aL = &sm.A[lm * A_STRIDE + hi * 16];
  // B fragment: lane n=lm holds column t, K k..k+15 (lanes 0-15) or
  // k+16..k+31 (lanes 16-31) -> 16 contiguous halves of vn row t.
  const size_t vbase = ((size_t)b * T_SZ + tbase + lm) * D_SZ + hi * 16;
  const _Float16* br0 = vnh + vbase;
  const _Float16* br1 = br0 + (size_t)16 * D_SZ;
  const _Float16* br2 = br0 + (size_t)32 * D_SZ;
  const _Float16* br3 = br0 + (size_t)48 * D_SZ;

  v8f acc0 = {}, acc1 = {}, acc2 = {}, acc3 = {};

  for (int k = 0; k < D_SZ; k += 32) {
    Frag A, B0, B1, B2, B3;                      // distinct regs -> load clustering
    A.h[0]  = *(const v8h*)(aL + k);
    A.h[1]  = *(const v8h*)(aL + k + 8);
    B0.h[0] = *(const v8h*)(br0 + k);
    B0.h[1] = *(const v8h*)(br0 + k + 8);
    B1.h[0] = *(const v8h*)(br1 + k);
    B1.h[1] = *(const v8h*)(br1 + k + 8);
    B2.h[0] = *(const v8h*)(br2 + k);
    B2.h[1] = *(const v8h*)(br2 + k + 8);
    B3.h[0] = *(const v8h*)(br3 + k);
    B3.h[1] = *(const v8h*)(br3 + k + 8);
    acc0 = __builtin_amdgcn_wmma_f32_16x16x32_f16(false, A.v, false, B0.v,
                                                  (short)0, acc0, false, false);
    acc1 = __builtin_amdgcn_wmma_f32_16x16x32_f16(false, A.v, false, B1.v,
                                                  (short)0, acc1, false, false);
    acc2 = __builtin_amdgcn_wmma_f32_16x16x32_f16(false, A.v, false, B2.v,
                                                  (short)0, acc2, false, false);
    acc3 = __builtin_amdgcn_wmma_f32_16x16x32_f16(false, A.v, false, B3.v,
                                                  (short)0, acc3, false, false);
  }

  // ---- Stage C tile in LDS for block-wide softmax over t --------------------
#pragma unroll
  for (int r = 0; r < 8; ++r) {
    float* o = &sm.S[(size_t)(r + hi * 8) * S_STRIDE + tbase + lm];
    o[0]  = acc0[r];
    o[16] = acc1[r];
    o[32] = acc2[r];
    o[48] = acc3[r];
  }
  __syncthreads();

  // ---- Masked softmax + argmax (rows jl = 2*wave, 2*wave+1) -----------------
  const int len = lengths[b];
  const float NEG = -__builtin_inff();
#pragma unroll
  for (int rr = 0; rr < 2; ++rr) {
    const int jl = wave * 2 + rr;
    const int j  = jbase + jl;
    const float4* srow = (const float4*)&sm.S[(size_t)jl * S_STRIDE];

    float4 x[4];
    float vmax = NEG, bestv = NEG;
    int   besti = T_SZ;
#pragma unroll
    for (int i = 0; i < 4; ++i) {
      const int t0 = lane * 4 + i * 128;
      float4 v = srow[lane + i * 32];
      v.x = (t0 + 0 < len) ? v.x : NEG;
      v.y = (t0 + 1 < len) ? v.y : NEG;
      v.z = (t0 + 2 < len) ? v.z : NEG;
      v.w = (t0 + 3 < len) ? v.w : NEG;
      x[i] = v;
      if (v.x > bestv) { bestv = v.x; besti = t0 + 0; }   // ascending t, strict '>'
      if (v.y > bestv) { bestv = v.y; besti = t0 + 1; }   // -> lowest index on tie
      if (v.z > bestv) { bestv = v.z; besti = t0 + 2; }
      if (v.w > bestv) { bestv = v.w; besti = t0 + 3; }
      vmax = fmaxf(vmax, fmaxf(fmaxf(v.x, v.y), fmaxf(v.z, v.w)));
    }
#pragma unroll
    for (int off = 16; off >= 1; off >>= 1) {
      vmax = fmaxf(vmax, __shfl_xor(vmax, off, 32));
      const float ov = __shfl_xor(bestv, off, 32);
      const int   oi = __shfl_xor(besti, off, 32);
      if (ov > bestv || (ov == bestv && oi < besti)) { bestv = ov; besti = oi; }
    }

    float s = 0.f;
#pragma unroll
    for (int i = 0; i < 4; ++i) {
      x[i].x = __expf(x[i].x - vmax);   // exp(-inf)=0 for masked frames
      x[i].y = __expf(x[i].y - vmax);
      x[i].z = __expf(x[i].z - vmax);
      x[i].w = __expf(x[i].w - vmax);
      s += x[i].x + x[i].y + x[i].z + x[i].w;
    }
#pragma unroll
    for (int off = 16; off >= 1; off >>= 1)
      s += __shfl_xor(s, off, 32);
    const float inv = 1.0f / s;

    float4* orow = (float4*)(out + (size_t)b * (B_SZ * T_SZ) + (size_t)j * T_SZ);
#pragma unroll
    for (int i = 0; i < 4; ++i) {
      float4 y = x[i];
      y.x *= inv; y.y *= inv; y.z *= inv; y.w *= inv;
      orow[lane + i * 32] = y;
    }

    // Diagonal: gather ORIGINAL video row at argmax frame -> video_max[b]
    if (j == b) {
      const float4* gsrc = (const float4*)(video + ((size_t)b * T_SZ + besti) * D_SZ);
      float4* gdst = (float4*)(out + (size_t)B_SZ * B_SZ * T_SZ + (size_t)b * D_SZ);
#pragma unroll
      for (int i = 0; i < 4; ++i)
        gdst[lane + i * 32] = gsrc[lane + i * 32];
    }
  }
}

// ---------------------------------------------------------------------------
extern "C" void kernel_launch(void* const* d_in, const int* in_sizes, int n_in,
                              void* d_out, int out_size, void* d_ws, size_t ws_size,
                              hipStream_t stream) {
  const float* video = (const float*)d_in[0];  // [256,512,512] f32
  const float* sent  = (const float*)d_in[1];  // [256,512]     f32
  const int*   lens  = (const int*)d_in[2];    // [256]         i32
  float* out = (float*)d_out;                  // sim [256,256,512] + video_max [256,512]

  _Float16* vnh = (_Float16*)d_ws;                       // 128 MB
  _Float16* snh = vnh + (size_t)B_SZ * T_SZ * D_SZ;      // +256 KB

  norm_rows_f16<<<(B_SZ * T_SZ) / 8, 256, 0, stream>>>(video, vnh, B_SZ * T_SZ);
  norm_rows_f16<<<B_SZ / 8, 256, 0, stream>>>(sent, snh, B_SZ);
  sim_fused_wmma<<<B_SZ * 16, 256, 0, stream>>>(snh, vnh, video, lens, out);
}
